// FoundationalTimeSeriesModel_31284541784600
// MI455X (gfx1250) — matrix operand order repaired and checked by reading, not compiled, not run on hardware
//
#include <hip/hip_runtime.h>
#include <hip/hip_bf16.h>
#include <math.h>

// ---------------------------------------------------------------------------
// CDNA5 / gfx1250 implementation. wave32. WMMA bf16 16x16x32, f32 accumulate.
// TDM (tensor_load_to_lds) stages conv weight tiles; s_wait_tensorcnt syncs.
// ---------------------------------------------------------------------------
typedef __bf16 bf16;
typedef __bf16 v16bf __attribute__((ext_vector_type(16)));
typedef float  v8f   __attribute__((ext_vector_type(8)));
typedef unsigned int v4ui __attribute__((ext_vector_type(4)));
typedef int v8i __attribute__((ext_vector_type(8)));
typedef int v4i __attribute__((ext_vector_type(4)));

#define WMMA_BF16(a, b, c) \
  __builtin_amdgcn_wmma_f32_16x16x32_bf16(false, (a), false, (b), (short)0, (c), false, false)

#if defined(__has_builtin)
#if __has_builtin(__builtin_amdgcn_tensor_load_to_lds) && \
    __has_builtin(__builtin_amdgcn_s_wait_tensorcnt)
#define HAS_TDM 1
#endif
#endif
#ifndef HAS_TDM
#define HAS_TDM 0
#endif

// Model constants
#define BN    32
#define TT    512
#define SS    32
#define NROW  1024        // B*S rows
#define T1    513         // T + 1 (cls)
#define CT    64          // PROJ == TCN_OUT
#define DM    128
#define NHEAD 8
#define HD    16
#define EXPT  16
#define MOEH  512
#define NEGF  (-3.3e38f)
#define WELEM (3 * 64 * 64)   // bf16 elems per conv weight tile (24KB)

__device__ __forceinline__ float bf2f(bf16 x) { return (float)x; }
__device__ __forceinline__ bf16  f2bf(float x) { return (bf16)x; }

// wave32 butterfly sum (lowers to ds_swizzle / v_permlane; no barriers)
__device__ __forceinline__ float wave_sum(float v) {
#pragma unroll
  for (int m = 16; m >= 1; m >>= 1) v += __shfl_xor(v, m, 32);
  return v;
}

// ---------------------------------------------------------------------------
// WMMA fragment layouts (CDNA5 ISA 7.12.2):
// A (bf16 16x32): lane<16: row M=lane, K {0..7}u{16..23}; lane>=16: row M=lane-16,
//   K {8..15}u{24..31}; element j: p=j>>1,q=j&1 -> K = p<4 ? hi*8+2p+q : 16+hi*8+2(p-4)+q
// B (bf16 32x16): lane: col N=lane&15; element j: K map as for A (code loads A and B alike with K = hi*16+j)
// C/D (f32 16x16): lane: N=lane&15; vgpr r: M = r + (lane>=16 ? 8 : 0)
// ---------------------------------------------------------------------------
__device__ __forceinline__ int a_kmap(int j, int hi) {
  int p = j >> 1, q = j & 1;
  return (p < 4) ? (hi * 8 + 2 * p + q) : (16 + hi * 8 + 2 * (p - 4) + q);
}

// ---------------------------------------------------------------------------
// Convert one conv weight tensor f32 (64co,64ci,3) -> bf16 slab [k][co][ci]
// ---------------------------------------------------------------------------
__global__ void k_w2bf(const float* __restrict__ w, bf16* __restrict__ o) {
  int i = blockIdx.x * blockDim.x + threadIdx.x;
  if (i >= WELEM) return;
  int k = i >> 12, r = i & 4095, co = r >> 6, ci = r & 63;
  o[i] = f2bf(w[(co * 64 + ci) * 3 + k]);
}

// ---------------------------------------------------------------------------
// input projection + cls + sinusoidal PE -> h0 bf16 [n][c][t]
// ---------------------------------------------------------------------------
__global__ void k_in_proj(const float* __restrict__ X, const float* __restrict__ w,
                          const float* __restrict__ b, const float* __restrict__ cls,
                          bf16* __restrict__ h) {
  int idx = blockIdx.x * blockDim.x + threadIdx.x;
  int total = NROW * CT * T1;
  if (idx >= total) return;
  int t = idx % T1;
  int c = (idx / T1) % CT;
  int n = idx / (T1 * CT);
  int i2 = c & ~1;
  float div = expf((float)i2 * (-logf(10000.0f) / (float)CT));
  float ang = (float)t * div;
  float pe = (c & 1) ? cosf(ang) : sinf(ang);
  float v;
  if (t == 0) {
    v = cls[c] + pe;
  } else {
    float xv = X[(size_t)n * TT + (t - 1)];  // x_features.reshape(B*S, T)
    v = xv * w[c] + b[c] + pe;
  }
  h[idx] = f2bf(v);
}

// ---------------------------------------------------------------------------
// dilated causal conv (64->64, KS=3) as WMMA GEMM. Weights arrive via TDM.
// x,y: bf16 [n][c][t]. mode 0: y=relu(conv+b); mode 1: y=relu(relu(conv+b)+resid)
// block 256 thr (8 waves), tile 64co x 64t; grid (tTiles, NROW)
// ---------------------------------------------------------------------------
#define XSTRIDE 80
__global__ void __launch_bounds__(256)
k_conv_wmma(const bf16* __restrict__ x, const bf16* __restrict__ wbf,
            const float* __restrict__ bias, const bf16* __restrict__ resid,
            bf16* __restrict__ y, int d, int mode) {
  __shared__ bf16 wlds[WELEM];              // [tap][co][ci]
  __shared__ bf16 xlds[64 * XSTRIDE];       // [ci][tt], tt = t - (t0-2d)

  const int tid  = threadIdx.x;
  const int lane = tid & 31;
  const int wv   = tid >> 5;
  const int hi   = (lane >> 4) & 1;
  const int l15  = lane & 15;
  const int n    = blockIdx.y;
  const int t0   = blockIdx.x * 64;
  const int tW   = 64 + 2 * d;

#if HAS_TDM
  // Tensor Data Mover: one DMA of the 24KB bf16 weight tile into LDS.
  // D# per ISA 8.3/8.4: count=1, type=2 (image), data_size=2B,
  // tensor_dim0 = tile_dim0 = tensor_dim0_stride = WELEM, 1 row.
  if (wv == 0) {
    unsigned long long ga = (unsigned long long)(size_t)wbf;
    unsigned int laddr = (unsigned int)(size_t)&wlds[0];
    v4ui g0;
    g0[0] = 1u;                                       // count=1, user mode
    g0[1] = laddr;                                    // lds_addr
    g0[2] = (unsigned int)(ga & 0xffffffffu);         // global_addr[31:0]
    g0[3] = (unsigned int)((ga >> 32) & 0x01ffffffu)  // global_addr[56:32]
            | (2u << 30);                             // type=2
    v8i g1;
    g1[0] = 0x00010000;                 // data_size=1 (2 bytes)
    g1[1] = (int)((WELEM & 0xffff) << 16);            // tensor_dim0[15:0]
    g1[2] = (int)((WELEM >> 16) | (1u << 16));        // dim0 hi | tensor_dim1=1
    g1[3] = (int)((WELEM & 0xffff) << 16);            // tile_dim0
    g1[4] = 1;                                        // tile_dim1=1, tile_dim2=0
    g1[5] = WELEM;                                    // tensor_dim0_stride lo32
    g1[6] = 0;
    g1[7] = 0;
    v4i gz; gz[0] = 0; gz[1] = 0; gz[2] = 0; gz[3] = 0;
#if __clang_major__ >= 23
    v8i gz8; for (int q = 0; q < 8; ++q) gz8[q] = 0;
    __builtin_amdgcn_tensor_load_to_lds(g0, g1, gz, gz, gz8, 0);
#else
    __builtin_amdgcn_tensor_load_to_lds(g0, g1, gz, gz, 0);
#endif
    __builtin_amdgcn_s_wait_tensorcnt(0);
  }
#else
  for (int i = tid; i < WELEM; i += 256) wlds[i] = wbf[i];
#endif

  // stage input tile with causal left halo (zero padded)
  for (int i = tid; i < 64 * tW; i += 256) {
    int ci = i / tW, tt = i - ci * tW;
    int gt = t0 - 2 * d + tt;
    bf16 v = (bf16)0.0f;
    if (gt >= 0 && gt < T1) v = x[((size_t)n * 64 + ci) * T1 + gt];
    xlds[ci * XSTRIDE + tt] = v;
  }
  __syncthreads();

  const int cobase = (wv >> 1) * 16;
  const int jt0    = (wv & 1) * 2;

  v8f acc[2];
#pragma unroll
  for (int r = 0; r < 8; ++r) { acc[0][r] = 0.0f; acc[1][r] = 0.0f; }

#pragma unroll
  for (int k = 0; k < 3; ++k) {
#pragma unroll
    for (int kc = 0; kc < 2; ++kc) {
      const int kbase = kc * 32;
      v16bf a;
#pragma unroll
      for (int j = 0; j < 16; ++j)
        a[j] = wlds[k * 4096 + (cobase + l15) * 64 + kbase + hi * 16 + j];
#pragma unroll
      for (int p = 0; p < 2; ++p) {
        const int jcol = (jt0 + p) * 16 + l15;   // output t within tile
        v16bf bfr;
#pragma unroll
        for (int j = 0; j < 16; ++j)
          bfr[j] = xlds[(kbase + hi * 16 + j) * XSTRIDE + jcol + k * d];
        acc[p] = WMMA_BF16(a, bfr, acc[p]);
      }
    }
  }

#pragma unroll
  for (int p = 0; p < 2; ++p) {
    const int t = t0 + (jt0 + p) * 16 + l15;
    if (t >= T1) continue;
#pragma unroll
    for (int r = 0; r < 8; ++r) {
      int co = cobase + r + hi * 8;
      size_t idx = ((size_t)n * 64 + co) * T1 + t;
      float v = acc[p][r] + bias[co];
      v = fmaxf(v, 0.0f);
      if (mode == 1) v = fmaxf(v + bf2f(resid[idx]), 0.0f);
      y[idx] = f2bf(v);
    }
  }
}

// ---------------------------------------------------------------------------
// Generic WMMA GEMM: C[M,N] = A[M,K](bf16) @ W[N,K]^T(f32) + bias
// optional relu, optional f32 residual; f32 and/or bf16 outputs (nullable).
// ---------------------------------------------------------------------------
#define KST 36
__global__ void __launch_bounds__(256)
k_gemm_wmma(const bf16* __restrict__ A, const float* __restrict__ W,
            const float* __restrict__ bias, const float* __restrict__ resid,
            float* __restrict__ outF, bf16* __restrict__ outB,
            int M, int N, int K, int relu) {
  __shared__ bf16 aS[64 * KST];
  __shared__ bf16 wS[64 * KST];

  const int tid  = threadIdx.x;
  const int lane = tid & 31;
  const int wv   = tid >> 5;
  const int hi   = (lane >> 4) & 1;
  const int l15  = lane & 15;
  const int m0   = blockIdx.y * 64;
  const int n0   = blockIdx.x * 64;
  const int rbase = (wv >> 1) * 16;
  const int cbase = (wv & 1) * 32;

  v8f acc[2];
#pragma unroll
  for (int r = 0; r < 8; ++r) { acc[0][r] = 0.0f; acc[1][r] = 0.0f; }

  for (int k0 = 0; k0 < K; k0 += 32) {
    for (int i = tid; i < 64 * 32; i += 256) {
      int r = i >> 5, kk = i & 31;
      int gk = k0 + kk;
      int gm = m0 + r;
      aS[r * KST + kk] = (gm < M && gk < K) ? A[(size_t)gm * K + gk] : (bf16)0.0f;
      int gn = n0 + r;
      wS[r * KST + kk] = (gn < N && gk < K) ? f2bf(W[(size_t)gn * K + gk]) : (bf16)0.0f;
    }
    if (k0 + 32 < K) {  // global_prefetch_b8 hint for next K chunk
      int pr = m0 + (tid & 63);
      if (pr < M) __builtin_prefetch(&A[(size_t)pr * K + k0 + 32], 0, 1);
      int pn = n0 + (tid & 63);
      if (pn < N) __builtin_prefetch(&W[(size_t)pn * K + k0 + 32], 0, 1);
    }
    __syncthreads();

    v16bf a;
#pragma unroll
    for (int j = 0; j < 16; ++j)
      a[j] = aS[(rbase + l15) * KST + hi * 16 + j];
#pragma unroll
    for (int p = 0; p < 2; ++p) {
      v16bf bfr;
#pragma unroll
      for (int j = 0; j < 16; ++j)
        bfr[j] = wS[(cbase + p * 16 + l15) * KST + hi * 16 + j];
      acc[p] = WMMA_BF16(a, bfr, acc[p]);
    }
    __syncthreads();
  }

#pragma unroll
  for (int p = 0; p < 2; ++p) {
    int nn = n0 + cbase + p * 16 + l15;
    if (nn >= N) continue;
    float bv = bias ? bias[nn] : 0.0f;
#pragma unroll
    for (int r = 0; r < 8; ++r) {
      int mm = m0 + rbase + r + hi * 8;
      if (mm >= M) continue;
      float v = acc[p][r] + bv;
      if (relu) v = fmaxf(v, 0.0f);
      if (resid) v += resid[(size_t)mm * N + nn];
      if (outF) outF[(size_t)mm * N + nn] = v;
      if (outB) outB[(size_t)mm * N + nn] = f2bf(v);
    }
  }
}

// ---------------------------------------------------------------------------
// TCN layernorm + pooling: per row n, LN over 64 ch at each t (2 waves).
// t=0 -> hcls; t in [1..512] -> masked mean pool + last position.
// ---------------------------------------------------------------------------
__global__ void k_tcn_norm_pool(const bf16* __restrict__ h, const float* __restrict__ g,
                                const float* __restrict__ b, const float* __restrict__ mask,
                                float* __restrict__ hcls, float* __restrict__ pooledF,
                                bf16* __restrict__ pooledB, float* __restrict__ lastF) {
  __shared__ float p1[2], p2[2];
  const int n = blockIdx.x;
  const int c = threadIdx.x;          // 64 threads = 2 waves
  const int wv = c >> 5;
  const float msk = mask[n];
  const bf16* row = h + ((size_t)n * 64 + c) * T1;
  float accu = 0.0f, lastv = 0.0f;
  for (int t = 0; t < T1; ++t) {
    float v = bf2f(row[t]);
    float s = wave_sum(v);
    float s2 = wave_sum(v * v);
    if ((c & 31) == 0) { p1[wv] = s; p2[wv] = s2; }
    __syncthreads();
    float m   = (p1[0] + p1[1]) * (1.0f / 64.0f);
    float var = (p2[0] + p2[1]) * (1.0f / 64.0f) - m * m;
    __syncthreads();
    float nv = (v - m) * rsqrtf(var + 1e-5f) * g[c] + b[c];
    if (t == 0) {
      hcls[n * 64 + c] = nv;
    } else {
      accu += nv;
      if (t == T1 - 1) lastv = nv;
    }
  }
  float pm = accu * (1.0f / 512.0f) * msk;
  pooledF[n * 64 + c] = pm;
  pooledB[n * 64 + c] = f2bf(pm);
  lastF[n * 64 + c]   = lastv * msk;
}

// ---------------------------------------------------------------------------
// Row LayerNorm (D == blockDim.x == 128 => 4 waves), f32/bf16 outs (nullable)
// ---------------------------------------------------------------------------
__global__ void k_ln_rows(const float* __restrict__ in, const float* __restrict__ g,
                          const float* __restrict__ b, int D,
                          float* __restrict__ outF, bf16* __restrict__ outB) {
  __shared__ float p1[4], p2[4];
  const int row = blockIdx.x;
  const int c = threadIdx.x;
  const int wv = c >> 5;
  float v = in[(size_t)row * D + c];
  float s = wave_sum(v);
  float s2 = wave_sum(v * v);
  if ((c & 31) == 0) { p1[wv] = s; p2[wv] = s2; }
  __syncthreads();
  float sum = p1[0] + p1[1] + p1[2] + p1[3];
  float sq  = p2[0] + p2[1] + p2[2] + p2[3];
  float m   = sum / (float)D;
  float var = sq / (float)D - m * m;
  float nv = (v - m) * rsqrtf(var + 1e-5f) * g[c] + b[c];
  if (outF) outF[(size_t)row * D + c] = nv;
  if (outB) outB[(size_t)row * D + c] = f2bf(nv);
}

// z = proj + pos_inter[s]
__global__ void k_z_init(const float* __restrict__ proj, const float* __restrict__ pos,
                         float* __restrict__ z) {
  int idx = blockIdx.x * blockDim.x + threadIdx.x;
  if (idx >= NROW * DM) return;
  int c = idx & (DM - 1);
  int n = idx >> 7;
  z[idx] = proj[idx] + pos[(n & 31) * DM + c];
}

// ---------------------------------------------------------------------------
// Attention over sensors: one wave per (b, head). S=32, HD=16.
// ---------------------------------------------------------------------------
__global__ void k_attn(const float* __restrict__ qkv, const float* __restrict__ mask,
                       bf16* __restrict__ oB) {
  __shared__ float kS[32][HD];
  __shared__ float vS[32][HD];
  const int bh = blockIdx.x;
  const int b = bh >> 3, h = bh & 7;
  const int s = threadIdx.x;  // wave32: lane == query index
  const int n = b * 32 + s;
  float q[HD];
#pragma unroll
  for (int d = 0; d < HD; ++d) {
    q[d]      = qkv[(size_t)n * 384 + h * HD + d];
    kS[s][d]  = qkv[(size_t)n * 384 + 128 + h * HD + d];
    vS[s][d]  = qkv[(size_t)n * 384 + 256 + h * HD + d];
  }
  __syncthreads();
  float sc[32];
  float mx = NEGF;
  for (int j = 0; j < 32; ++j) {
    float t = 0.0f;
#pragma unroll
    for (int d = 0; d < HD; ++d) t += q[d] * kS[j][d];
    t *= 0.25f;  // 1/sqrt(16)
    if (mask[b * 32 + j] == 0.0f) t = NEGF;
    sc[j] = t;
    mx = fmaxf(mx, t);
  }
  float den = 0.0f;
  for (int j = 0; j < 32; ++j) { sc[j] = expf(sc[j] - mx); den += sc[j]; }
  float inv = 1.0f / den;
#pragma unroll
  for (int d = 0; d < HD; ++d) {
    float o = 0.0f;
    for (int j = 0; j < 32; ++j) o += sc[j] * vS[j][d];
    oB[(size_t)n * DM + h * HD + d] = f2bf(o * inv);
  }
}

// ctx_m = ctx * mask  (f32 + bf16)
__global__ void k_ctx_mask(const float* __restrict__ ctx, const float* __restrict__ mask,
                           float* __restrict__ cmF, bf16* __restrict__ cmB) {
  int idx = blockIdx.x * blockDim.x + threadIdx.x;
  if (idx >= NROW * DM) return;
  int n = idx >> 7;
  float v = ctx[idx] * mask[n];
  cmF[idx] = v;
  cmB[idx] = f2bf(v);
}

// pooled context + gate input assembly. grid=B, block=128.
__global__ void k_pool_ctx(const float* __restrict__ ctx, const float* __restrict__ cmF,
                           const float* __restrict__ hcls, const float* __restrict__ mask,
                           float* __restrict__ g_in, bf16* __restrict__ meanB) {
  const int b = blockIdx.x;
  const int c = threadIdx.x;
  float active = 0.0f;
  for (int s = 0; s < 32; ++s) active += mask[b * 32 + s];
  active = fmaxf(active, 1.0f);
  float mean = 0.0f, mx = NEGF;
  for (int s = 0; s < 32; ++s) {
    int n = b * 32 + s;
    mean += cmF[(size_t)n * DM + c];
    float cv = (mask[n] != 0.0f) ? ctx[(size_t)n * DM + c] : NEGF;
    mx = fmaxf(mx, cv);
  }
  mean /= active;
  g_in[b * 320 + c] = mean;
  g_in[b * 320 + 128 + c] = mx;
  meanB[b * DM + c] = f2bf(mean);
  if (c < 64) {
    float mh = 0.0f;
    for (int s = 0; s < 32; ++s) mh += hcls[(b * 32 + s) * 64 + c] * mask[b * 32 + s];
    g_in[b * 320 + 256 + c] = mh / active;
  }
}

// ---------------------------------------------------------------------------
// Top-2 gating + aux accumulators. One thread per row.
// acc: [0..15]=sum probs, [16..31]=topk counts, [32]=entropy sum
// ---------------------------------------------------------------------------
__global__ void k_gate_topk(const float* __restrict__ G, int Kg,
                            const float* __restrict__ gw, const float* __restrict__ gb,
                            int Nrows, int* __restrict__ topi, float* __restrict__ topw,
                            float* __restrict__ acc) {
  int row = blockIdx.x * blockDim.x + threadIdx.x;
  if (row >= Nrows) return;
  float lg[EXPT];
  for (int e = 0; e < EXPT; ++e) {
    float t = gb[e];
    const float* wr = gw + (size_t)e * Kg;
    const float* gr = G + (size_t)row * Kg;
    for (int k = 0; k < Kg; ++k) t += wr[k] * gr[k];
    lg[e] = t;
  }
  float v0 = NEGF, v1 = NEGF; int i0 = 0, i1 = 0;
  for (int e = 0; e < EXPT; ++e) {
    if (lg[e] > v0) { v1 = v0; i1 = i0; v0 = lg[e]; i0 = e; }
    else if (lg[e] > v1) { v1 = lg[e]; i1 = e; }
  }
  float e1 = expf(v1 - v0);
  float w0 = 1.0f / (1.0f + e1);
  float w1 = e1 / (1.0f + e1);
  float se = 0.0f;
  for (int e = 0; e < EXPT; ++e) se += expf(lg[e] - v0);
  float lse = v0 + logf(se);
  float ent = 0.0f;
  for (int e = 0; e < EXPT; ++e) {
    float lp = lg[e] - lse;
    float pp = expf(lp);
    ent -= pp * lp;
    atomicAdd(&acc[e], pp);
  }
  atomicAdd(&acc[16 + i0], 1.0f);
  atomicAdd(&acc[16 + i1], 1.0f);
  atomicAdd(&acc[32], ent);
  topi[row * 2] = i0; topi[row * 2 + 1] = i1;
  topw[row * 2] = w0; topw[row * 2 + 1] = w1;
}

// y[n] = sum_k topw[n,k] * all_out[topi[n,k]][n]
__global__ void k_moe_select(const float* __restrict__ allout, int Nrows,
                             const int* __restrict__ topi, const float* __restrict__ topw,
                             float* __restrict__ y) {
  int idx = blockIdx.x * blockDim.x + threadIdx.x;
  if (idx >= Nrows * DM) return;
  int o = idx & (DM - 1);
  int n = idx >> 7;
  float v = topw[n * 2] *     allout[((size_t)topi[n * 2]     * Nrows + n) * DM + o] +
            topw[n * 2 + 1] * allout[((size_t)topi[n * 2 + 1] * Nrows + n) * DM + o];
  y[idx] = v;
}

// prediction head: concat(last_tcn, ctx_m, y_fc[b]) @ pred_w^T + pred_b
__global__ void k_pred(const float* __restrict__ lastT, const float* __restrict__ cmF,
                       const float* __restrict__ yfc, const float* __restrict__ pw,
                       const float* __restrict__ pb, float* __restrict__ out) {
  int idx = blockIdx.x * blockDim.x + threadIdx.x;
  if (idx >= NROW * 3) return;
  int j = idx % 3, n = idx / 3, b = n >> 5;
  const float* wr = pw + j * 320;
  float t = pb[j];
  for (int c = 0; c < 64; ++c)  t += wr[c]       * lastT[n * 64 + c];
  for (int c = 0; c < 128; ++c) t += wr[64 + c]  * cmF[(size_t)n * DM + c];
  for (int c = 0; c < 128; ++c) t += wr[192 + c] * yfc[b * DM + c];
  out[idx] = t;
}

__global__ void k_fail(const float* __restrict__ yfa, const float* __restrict__ fw,
                       const float* __restrict__ fb, float* __restrict__ out) {
  int idx = threadIdx.x;
  if (idx >= BN * 3) return;
  int j = idx % 3, b = idx / 3;
  float t = fb[j];
  for (int c = 0; c < 128; ++c) t += fw[j * 128 + c] * yfa[b * DM + c];
  out[idx] = t;
}

__global__ void k_rca(const float* __restrict__ lastT, const float* __restrict__ cmF,
                      const float* __restrict__ yrc, const float* __restrict__ X,
                      const float* __restrict__ lastK, const float* __restrict__ rw,
                      const float* __restrict__ rb, float* __restrict__ out) {
  int n = blockIdx.x * blockDim.x + threadIdx.x;
  if (n >= NROW) return;
  int b = n >> 5, s = n & 31;
  float nov = fabsf(X[(size_t)b * TT * SS + 511 * SS + s] - lastK[b * 32 + s]);
  float t = rb[0];
  for (int c = 0; c < 64; ++c)  t += rw[c]       * lastT[n * 64 + c];
  for (int c = 0; c < 128; ++c) t += rw[64 + c]  * cmF[(size_t)n * DM + c];
  for (int c = 0; c < 128; ++c) t += rw[192 + c] * yrc[(size_t)n * DM + c];
  t += rw[320] * nov;
  out[n] = t;
}

// aux = 0.01*(lb_fc+lb_fa+lb_rc) - 0.001*(ent_fc+ent_fa+ent_rc)
__global__ void k_aux(const float* __restrict__ acc, float* __restrict__ out) {
  if (threadIdx.x != 0) return;
  float lb = 0.0f, ent = 0.0f;
  for (int gi = 0; gi < 3; ++gi) {
    const float* a = acc + gi * 40;
    float N = (gi == 2) ? 1024.0f : 32.0f;
    float s = 0.0f;
    for (int e = 0; e < EXPT; ++e) s += (a[e] / N) * (a[16 + e] / (2.0f * N));
    lb += (float)EXPT * s;
    ent += a[32] / N;
  }
  out[0] = 0.01f * lb - 0.001f * ent;
}

__global__ void k_zero(float* __restrict__ p, int n) {
  int idx = blockIdx.x * blockDim.x + threadIdx.x;
  if (idx < n) p[idx] = 0.0f;
}

// ---------------------------------------------------------------------------
// Host side
// ---------------------------------------------------------------------------
static void gemm(const bf16* A, const float* W, const float* bias, const float* resid,
                 float* outF, bf16* outB, int M, int N, int K, int relu, hipStream_t s) {
  dim3 g((N + 63) / 64, (M + 63) / 64);
  k_gemm_wmma<<<g, 256, 0, s>>>(A, W, bias, resid, outF, outB, M, N, K, relu);
}

extern "C" void kernel_launch(void* const* d_in, const int* in_sizes, int n_in,
                              void* d_out, int out_size, void* d_ws, size_t ws_size,
                              hipStream_t stream) {
  auto in = [&](int i) { return (const float*)d_in[i]; };
  // inputs
  const float* X = in(0);
  const float* MASK = in(1);
  const float* LASTK = in(2);
  // params (jax pytree: dict keys sorted)
  const float* We1 = in(3);   const float* We2 = in(4);
  const float* be1 = in(5);   const float* be2 = in(6);
  const float* cls = in(7);
  const float* fail_b = in(8);  const float* fail_w = in(9);
  const float* gfa_b = in(10);  const float* gfa_w = in(11);
  const float* gfc_b = in(12);  const float* gfc_w = in(13);
  const float* grc_b = in(14);  const float* grc_w = in(15);
  const float* inp_b = in(16);  const float* inp_w = in(17);
  const float* pool_b = in(18); const float* pool_w = in(19);
  const float* pos = in(20);
  const float* pred_b = in(21); const float* pred_w = in(22);
  const float* rca_b = in(23);  const float* rca_w = in(24);
  // tcn: 25 + lvl*4 -> {b1,b2,w1,w2}
  const float* tnorm_b = in(41); const float* tnorm_g = in(42);
  // tr layer L: 43 + L*12 -> {ff1_b,ff1_w,ff2_b,ff2_w,ln1_b,ln1_g,ln2_b,ln2_g,out_b,out_w,qkv_b,qkv_w}
  const float* trn_b = in(67); const float* trn_g = in(68);

  float* OUT = (float*)d_out;

  // workspace bump allocator
  char* base = (char*)d_ws;
  size_t off = 0;
  auto alloc = [&](size_t bytes) -> void* {
    void* p = base + off;
    off = (off + bytes + 255) & ~(size_t)255;
    return p;
  };
  const size_t HN = (size_t)NROW * CT * T1;
  bf16* hA   = (bf16*)alloc(HN * 2);
  bf16* hT   = (bf16*)alloc(HN * 2);
  bf16* wbf  = (bf16*)alloc((size_t)8 * WELEM * 2);   // bf16 conv weight slabs
  float* pooledF = (float*)alloc(NROW * 64 * 4);
  bf16*  pooledB = (bf16*)alloc(NROW * 64 * 2);
  float* lastT   = (float*)alloc(NROW * 64 * 4);
  float* hcls    = (float*)alloc(NROW * 64 * 4);
  float* z    = (float*)alloc(NROW * DM * 4);
  bf16*  znB  = (bf16*)alloc(NROW * DM * 2);
  float* qkv  = (float*)alloc(NROW * 384 * 4);
  bf16*  attnB = (bf16*)alloc(NROW * DM * 2);
  bf16*  hffB  = (bf16*)alloc(NROW * 256 * 2);
  float* ctx   = (float*)alloc(NROW * DM * 4);
  float* cmF   = (float*)alloc(NROW * DM * 4);
  bf16*  cmB   = (bf16*)alloc(NROW * DM * 2);
  float* proj  = (float*)alloc(NROW * DM * 4);
  float* g_in  = (float*)alloc(BN * 320 * 4);
  bf16*  meanB = (bf16*)alloc(BN * DM * 2);
  bf16*  hMoe  = (bf16*)alloc((size_t)NROW * MOEH * 2);
  float* aoTok = (float*)alloc((size_t)EXPT * NROW * DM * 4);
  float* aoMean = (float*)alloc((size_t)EXPT * BN * DM * 4);
  float* yfc = (float*)alloc(BN * DM * 4);
  float* yfa = (float*)alloc(BN * DM * 4);
  float* yrc = (float*)alloc(NROW * DM * 4);
  int*   ti_fc = (int*)alloc(NROW * 2 * 4);  float* tw_fc = (float*)alloc(NROW * 2 * 4);
  int*   ti_fa = (int*)alloc(NROW * 2 * 4);  float* tw_fa = (float*)alloc(NROW * 2 * 4);
  int*   ti_rc = (int*)alloc(NROW * 2 * 4);  float* tw_rc = (float*)alloc(NROW * 2 * 4);
  float* gacc = (float*)alloc(3 * 40 * 4);
  (void)ws_size; (void)in_sizes; (void)n_in; (void)out_size;

  k_zero<<<1, 128, 0, stream>>>(gacc, 3 * 40);

  // ---- pre-convert conv weights to bf16 slabs ([k][co][ci]) ----
  for (int lvl = 0; lvl < 4; ++lvl) {
    const float* w1 = in(25 + lvl * 4 + 2);
    const float* w2 = in(25 + lvl * 4 + 3);
    k_w2bf<<<(WELEM + 255) / 256, 256, 0, stream>>>(w1, wbf + (size_t)(lvl * 2 + 0) * WELEM);
    k_w2bf<<<(WELEM + 255) / 256, 256, 0, stream>>>(w2, wbf + (size_t)(lvl * 2 + 1) * WELEM);
  }

  // ---- TCN front end ----
  {
    int total = NROW * CT * T1;
    k_in_proj<<<(total + 255) / 256, 256, 0, stream>>>(X, inp_w, inp_b, cls, hA);
  }
  dim3 cgrid((T1 + 63) / 64, NROW);
  for (int lvl = 0; lvl < 4; ++lvl) {
    int d = 1 << lvl;
    const float* b1 = in(25 + lvl * 4 + 0);
    const float* b2 = in(25 + lvl * 4 + 1);
    k_conv_wmma<<<cgrid, 256, 0, stream>>>(hA, wbf + (size_t)(lvl * 2 + 0) * WELEM,
                                           b1, nullptr, hT, d, 0);
    k_conv_wmma<<<cgrid, 256, 0, stream>>>(hT, wbf + (size_t)(lvl * 2 + 1) * WELEM,
                                           b2, hA, hA, d, 1);  // in-place resid ok
  }
  k_tcn_norm_pool<<<NROW, 64, 0, stream>>>(hA, tnorm_g, tnorm_b, MASK, hcls,
                                           pooledF, pooledB, lastT);

  // ---- inter-sensor transformer ----
  gemm(pooledB, pool_w, pool_b, nullptr, proj, nullptr, NROW, DM, 64, 0, stream);
  k_z_init<<<(NROW * DM + 255) / 256, 256, 0, stream>>>(proj, pos, z);
  for (int L = 0; L < 2; ++L) {
    int pbase = 43 + L * 12;
    const float* ff1_b = in(pbase + 0); const float* ff1_w = in(pbase + 1);
    const float* ff2_b = in(pbase + 2); const float* ff2_w = in(pbase + 3);
    const float* ln1_b = in(pbase + 4); const float* ln1_g = in(pbase + 5);
    const float* ln2_b = in(pbase + 6); const float* ln2_g = in(pbase + 7);
    const float* out_b = in(pbase + 8); const float* out_w = in(pbase + 9);
    const float* qkv_b = in(pbase + 10); const float* qkv_w = in(pbase + 11);
    k_ln_rows<<<NROW, 128, 0, stream>>>(z, ln1_g, ln1_b, DM, nullptr, znB);
    gemm(znB, qkv_w, qkv_b, nullptr, qkv, nullptr, NROW, 384, DM, 0, stream);
    k_attn<<<BN * NHEAD, 32, 0, stream>>>(qkv, MASK, attnB);
    gemm(attnB, out_w, out_b, z, z, nullptr, NROW, DM, DM, 0, stream);
    k_ln_rows<<<NROW, 128, 0, stream>>>(z, ln2_g, ln2_b, DM, nullptr, znB);
    gemm(znB, ff1_w, ff1_b, nullptr, nullptr, hffB, NROW, 256, DM, 1, stream);
    gemm(hffB, ff2_w, ff2_b, z, z, nullptr, NROW, DM, 256, 0, stream);
  }
  k_ln_rows<<<NROW, 128, 0, stream>>>(z, trn_g, trn_b, DM, ctx, nullptr);
  k_ctx_mask<<<(NROW * DM + 255) / 256, 256, 0, stream>>>(ctx, MASK, cmF, cmB);
  k_pool_ctx<<<BN, 128, 0, stream>>>(ctx, cmF, hcls, MASK, g_in, meanB);

  // ---- gating ----
  k_gate_topk<<<1, 32, 0, stream>>>(g_in, 320, gfc_w, gfc_b, BN, ti_fc, tw_fc, gacc + 0);
  k_gate_topk<<<1, 32, 0, stream>>>(g_in, 320, gfa_w, gfa_b, BN, ti_fa, tw_fa, gacc + 40);
  k_gate_topk<<<(NROW + 255) / 256, 256, 0, stream>>>(cmF, DM, grc_w, grc_b, NROW,
                                                      ti_rc, tw_rc, gacc + 80);

  // ---- MoE experts (dense all-expert compute, then top-2 select) ----
  for (int e = 0; e < EXPT; ++e) {
    const float* w1 = We1 + (size_t)e * MOEH * DM;
    const float* bb1 = be1 + (size_t)e * MOEH;
    const float* w2 = We2 + (size_t)e * DM * MOEH;
    const float* bb2 = be2 + (size_t)e * DM;
    gemm(meanB, w1, bb1, nullptr, nullptr, hMoe, BN, MOEH, DM, 1, stream);
    gemm(hMoe, w2, bb2, nullptr, aoMean + (size_t)e * BN * DM, nullptr, BN, DM, MOEH, 0, stream);
    gemm(cmB, w1, bb1, nullptr, nullptr, hMoe, NROW, MOEH, DM, 1, stream);
    gemm(hMoe, w2, bb2, nullptr, aoTok + (size_t)e * NROW * DM, nullptr, NROW, DM, MOEH, 0, stream);
  }
  k_moe_select<<<(BN * DM + 255) / 256, 256, 0, stream>>>(aoMean, BN, ti_fc, tw_fc, yfc);
  k_moe_select<<<(BN * DM + 255) / 256, 256, 0, stream>>>(aoMean, BN, ti_fa, tw_fa, yfa);
  k_moe_select<<<(NROW * DM + 255) / 256, 256, 0, stream>>>(aoTok, NROW, ti_rc, tw_rc, yrc);

  // ---- output heads ----
  k_pred<<<(NROW * 3 + 255) / 256, 256, 0, stream>>>(lastT, cmF, yfc, pred_w, pred_b, OUT);
  k_fail<<<1, 96, 0, stream>>>(yfa, fail_w, fail_b, OUT + 3072);
  k_rca<<<(NROW + 255) / 256, 256, 0, stream>>>(lastT, cmF, yrc, X, LASTK, rca_w, rca_b,
                                                OUT + 3168);
  k_aux<<<1, 32, 0, stream>>>(gacc, OUT + 4192);
}
